// UniformAttentionSublayer_84404697301732
// MI455X (gfx1250) — matrix operand order repaired and checked by reading, not compiled, not run on hardware
//
#include <hip/hip_runtime.h>

// ---------------------------------------------------------------------------
// Relative-position attention sublayer for MI455X (gfx1250, wave32, WMMA).
// All matmuls via v_wmma_f32_16x16x32_bf16; flash-style fused attention so the
// [B,H,S,S] logits tensor never touches HBM. LDS tiles are filled with the
// CDNA5 async paths (TDM tensor_load_to_lds / global_load_async_to_lds_b128)
// when the toolchain exposes them, double-buffered to overlap DMA with WMMA.
// ---------------------------------------------------------------------------

#define B_DIM 4
#define S_DIM 2048
#define D_DIM 1024
#define H_DIM 16
#define DH    64
#define NPOS  257
#define NPOSP 272   // 257 padded up to a multiple of 16

typedef __attribute__((ext_vector_type(16))) __bf16         bf16x16;
typedef __attribute__((ext_vector_type(16))) unsigned short ushx16;
typedef __attribute__((ext_vector_type(8)))  float          floatx8;
typedef __attribute__((ext_vector_type(4)))  unsigned int   uintx4;
typedef __attribute__((ext_vector_type(8)))  int            intx8;
typedef __attribute__((ext_vector_type(4)))  int            intx4;

// ---- CDNA5 feature detection ----------------------------------------------
#if defined(__has_builtin)
#if __has_builtin(__builtin_amdgcn_global_load_async_to_lds_b128) && \
    __has_builtin(__builtin_amdgcn_s_wait_asynccnt)
#define HAVE_ASYNC 1
#endif
#if __has_builtin(__builtin_amdgcn_tensor_load_to_lds) && \
    __has_builtin(__builtin_amdgcn_s_wait_tensorcnt)
#define HAVE_TDM 1
#if __has_include(<hip/amd_detail/amd_gfx1250_TDM.h>)
#define TDM_6ARG 1
#endif
#endif
#endif
#ifndef HAVE_ASYNC
#define HAVE_ASYNC 0
#endif
#ifndef HAVE_TDM
#define HAVE_TDM 0
#endif

#if HAVE_ASYNC
// Builtin signature (per hipcc diagnostic): (int4 AS1*, int4 AS3*, imm, imm)
#define TILE_CP(dst, src)                                                   \
  __builtin_amdgcn_global_load_async_to_lds_b128(                           \
      (__attribute__((address_space(1))) intx4*)(const void*)(src),         \
      (__attribute__((address_space(3))) intx4*)(void*)(dst), 0, 0)
#define ASYNC_WAIT(n) __builtin_amdgcn_s_wait_asynccnt(n)
#else
#define TILE_CP(dst, src) (*(uint4*)(dst) = *(const uint4*)(src))
#define ASYNC_WAIT(n)
#endif

union FragU { ushx16 u; bf16x16 b; };

__device__ __forceinline__ unsigned short f2bf(float f) {
  union { float f; unsigned int u; } x; x.f = f;
  unsigned int r = x.u + 0x7FFFu + ((x.u >> 16) & 1u);   // round-to-nearest-even
  return (unsigned short)(r >> 16);
}

__device__ __forceinline__ unsigned lds_offset_of(const void* p) {
  return (unsigned)(unsigned long long)
      (__attribute__((address_space(3))) const void*)p;
}

#if HAVE_TDM
// 2-D bf16 tile DMA: tile_d1 rows x tile_d0 elems (2B), row stride stride0
// elems, packed contiguously at LDS byte offset lds_off.
__device__ __forceinline__ void tdm_load_2d(unsigned lds_off, const void* gsrc,
                                            unsigned tile_d0, unsigned tile_d1,
                                            unsigned tensor_d0, unsigned tensor_d1,
                                            unsigned long long stride0) {
  unsigned long long ga = (unsigned long long)gsrc;
  uintx4 g0;
  g0[0] = 1u;                                          // count=1 (valid D#)
  g0[1] = lds_off;                                     // lds_addr
  g0[2] = (unsigned)(ga & 0xFFFFFFFFu);                // global_addr[31:0]
  g0[3] = (unsigned)((ga >> 32) & 0x1FFFFFFu) | (2u << 30);  // addr[56:32], type=2
  intx8 g1;
  g1[0] = (int)(1u << 16);                             // wg_mask=0, data_size=2B
  g1[1] = (int)((tensor_d0 & 0xFFFFu) << 16);          // tensor_dim0[15:0]
  g1[2] = (int)(((tensor_d0 >> 16) & 0xFFFFu) | ((tensor_d1 & 0xFFFFu) << 16));
  g1[3] = (int)(((tensor_d1 >> 16) & 0xFFFFu) | ((tile_d0 & 0xFFFFu) << 16));
  g1[4] = (int)(tile_d1 & 0xFFFFu);                    // tile_dim1, tile_dim2=0
  g1[5] = (int)(unsigned)(stride0 & 0xFFFFFFFFu);      // dim0_stride[31:0]
  g1[6] = (int)(unsigned)((stride0 >> 32) & 0xFFFFu);  // dim0_stride[47:32]
  g1[7] = 0;
  intx4 gz = (intx4)0;
#if TDM_6ARG
  intx8 gz8 = (intx8)0;
  __builtin_amdgcn_tensor_load_to_lds(g0, g1, gz, gz, gz8, 0);
#else
  __builtin_amdgcn_tensor_load_to_lds(g0, g1, gz, gz, 0);
#endif
}
#endif

// A fragment: 16(M) x 32(K) bf16, element (m,k) at p0[m*ld + k].
__device__ __forceinline__ bf16x16 load_a_frag(const unsigned short* p0, int ld, int lane) {
  const int m    = lane & 15;
  const int koff = (lane >> 4) << 4;       // 0 or 16
  const unsigned short* p = p0 + m * ld + koff;
  ushx16 a;
#pragma unroll
  for (int e = 0; e < 16; ++e) a[e] = p[e];         // K = koff + 0..15
  FragU f; f.u = a; return f.b;
}

// B fragment: 32(K) x 16(N) bf16, element (k,n) at p0[n*ld + k].
__device__ __forceinline__ bf16x16 load_b_frag(const unsigned short* p0, int ld, int lane) {
  const int n  = lane & 15;
  const int kb = (lane >> 4) << 4;         // 0 or 16
  const unsigned short* p = p0 + n * ld + kb;
  ushx16 b;
#pragma unroll
  for (int e = 0; e < 16; ++e) b[e] = p[e];
  FragU f; f.u = b; return f.b;
}

// B fragment where element (k,n) is at p0[k*ld + n]  (row-major K x N source).
__device__ __forceinline__ bf16x16 load_b_frag_t(const unsigned short* p0, int ld, int lane) {
  const int n  = lane & 15;
  const int kb = (lane >> 4) << 4;
  ushx16 b;
#pragma unroll
  for (int e = 0; e < 16; ++e) b[e] = p0[(kb + e) * ld + n];
  FragU f; f.u = b; return f.b;
}

__device__ __forceinline__ floatx8 wmma_bf16(bf16x16 a, bf16x16 b, floatx8 c) {
  return __builtin_amdgcn_wmma_f32_16x16x32_bf16(false, a, false, b, (short)0, c, false, false);
}

// ---------------------------------------------------------------------------
// fp32 -> bf16 casts
// ---------------------------------------------------------------------------
__global__ void cast_bf16_kernel(const float* __restrict__ src,
                                 unsigned short* __restrict__ dst, int n) {
  int i = blockIdx.x * blockDim.x + threadIdx.x;
  if (i < n) dst[i] = f2bf(src[i]);
}

__global__ void cast_pos_kernel(const float* __restrict__ src,
                                unsigned short* __restrict__ dst) {
  int i = blockIdx.x * blockDim.x + threadIdx.x;
  if (i >= NPOSP * DH) return;
  int row = i >> 6;
  dst[i] = (row < NPOS) ? f2bf(src[i]) : (unsigned short)0;
}

// ---------------------------------------------------------------------------
// Generic bf16 WMMA GEMM:  C[M,N] (f32 acc) = A[M,K] * Bm[N,K]^T (+ bias)
// Block tile 64x128, 8 waves of 32x32, K steps of 64, double-buffered LDS.
// M must be a multiple of 64 (true for all call sites); N edge handled by
// clamped loads + store guard. mode 0: fp32 store; mode 1: QKV bf16 scatter.
// ---------------------------------------------------------------------------
#define BM 64
#define BN 128
#define BK 64

__device__ __forceinline__ void gemm_issue_tile(const unsigned short* __restrict__ A,
                                                const unsigned short* __restrict__ Bm,
                                                unsigned short* As, unsigned short* Bs,
                                                int m0, int n0, int Ndim, int Kdim,
                                                int kt, int t) {
  // A tile 64x64: 512 x b128 chunks, 2 per thread (always in bounds)
#pragma unroll
  for (int i = 0; i < 2; ++i) {
    int chunk = t + i * 256;
    int r = chunk >> 3;
    int c = (chunk & 7) << 3;
    TILE_CP(As + r * BK + c, A + (size_t)(m0 + r) * Kdim + kt + c);
  }
  // B tile 128x64: 1024 chunks, 4 per thread; clamp OOB rows (results discarded)
#pragma unroll
  for (int i = 0; i < 4; ++i) {
    int chunk = t + i * 256;
    int r = chunk >> 3;
    int c = (chunk & 7) << 3;
    int gn = n0 + r;
    if (gn >= Ndim) gn = Ndim - 1;
    TILE_CP(Bs + r * BK + c, Bm + (size_t)gn * Kdim + kt + c);
  }
}

__global__ __launch_bounds__(256)
void gemm_bf16_wmma(const unsigned short* __restrict__ A,
                    const unsigned short* __restrict__ Bm,
                    const float* __restrict__ bias,
                    int Mdim, int Ndim, int Kdim,
                    int mode, float* __restrict__ Cf, int ldc,
                    unsigned short* __restrict__ Qb,
                    unsigned short* __restrict__ Kbp,
                    unsigned short* __restrict__ Vb) {
  __shared__ __align__(16) unsigned short As[2 * BM * BK];   // 16 KB
  __shared__ __align__(16) unsigned short Bs[2 * BN * BK];   // 32 KB
  const int t    = threadIdx.x;
  const int lane = t & 31;
  const int wave = t >> 5;
  const int wm   = wave >> 2;   // 0..1
  const int wn   = wave & 3;    // 0..3
  const int m0   = blockIdx.y * BM;
  const int n0   = blockIdx.x * BN;

  floatx8 acc[2][2];
#pragma unroll
  for (int i = 0; i < 2; ++i)
#pragma unroll
    for (int j = 0; j < 2; ++j) acc[i][j] = (floatx8)0.0f;

  const int T = Kdim >> 6;
  gemm_issue_tile(A, Bm, As, Bs, m0, n0, Ndim, Kdim, 0, t);

  for (int it = 0; it < T; ++it) {
    if (it + 1 < T) {
      gemm_issue_tile(A, Bm, As + ((it + 1) & 1) * (BM * BK),
                      Bs + ((it + 1) & 1) * (BN * BK),
                      m0, n0, Ndim, Kdim, (it + 1) << 6, t);
      ASYNC_WAIT(6);           // in-order: current tile resident
    } else {
      ASYNC_WAIT(0);
    }
    __syncthreads();

    const unsigned short* Ab = As + (it & 1) * (BM * BK);
    const unsigned short* Bb = Bs + (it & 1) * (BN * BK);
#pragma unroll
    for (int kk = 0; kk < 2; ++kk) {
      bf16x16 af[2], bfg[2];
#pragma unroll
      for (int i = 0; i < 2; ++i)
        af[i] = load_a_frag(Ab + (wm * 32 + i * 16) * BK + kk * 32, BK, lane);
#pragma unroll
      for (int j = 0; j < 2; ++j)
        bfg[j] = load_b_frag(Bb + (wn * 32 + j * 16) * BK + kk * 32, BK, lane);
#pragma unroll
      for (int i = 0; i < 2; ++i)
#pragma unroll
        for (int j = 0; j < 2; ++j)
          acc[i][j] = wmma_bf16(af[i], bfg[j], acc[i][j]);
    }
    __syncthreads();           // safe to overwrite this buffer next round
  }

  const int nl = lane & 15;
  const int mh = (lane >> 4) << 3;
#pragma unroll
  for (int i = 0; i < 2; ++i)
#pragma unroll
    for (int j = 0; j < 2; ++j)
#pragma unroll
      for (int r = 0; r < 8; ++r) {
        int gm = m0 + wm * 32 + i * 16 + r + mh;
        int gn = n0 + wn * 32 + j * 16 + nl;
        if (gm >= Mdim || gn >= Ndim) continue;
        float v = acc[i][j][r];
        if (bias) v += bias[gn];
        if (mode == 0) {
          Cf[(size_t)gm * ldc + gn] = v;
        } else {
          int bb = gm >> 11, ss = gm & (S_DIM - 1);
          int h = gn / 192, r2 = gn - h * 192;
          size_t base = (((size_t)bb * H_DIM + h) * S_DIM + ss) * DH;
          unsigned short bv = f2bf(v);
          if (r2 < 64)        Qb[base + r2]        = bv;
          else if (r2 < 128)  Kbp[base + r2 - 64]  = bv;
          else                Vb[base + r2 - 128]  = bv;
        }
      }
}

// ---------------------------------------------------------------------------
// Fused flash attention with relative-position logits.
// grid (B*H, S/128); block 256 = 8 waves, each wave owns 16 q-rows.
// K/V tiles (64 keys x 64 dh, bf16) double-buffered; TDM-driven when present.
// ---------------------------------------------------------------------------
__device__ __forceinline__ void kv_issue(const unsigned short* __restrict__ Kbase,
                                         const unsigned short* __restrict__ Vbase,
                                         unsigned short* KtBuf, unsigned short* VtBuf,
                                         int k0, int t) {
#if HAVE_TDM
  if (t < 32) {
    tdm_load_2d(lds_offset_of(KtBuf), Kbase + (size_t)k0 * DH, 64, 64, 64, 64, 64);
    tdm_load_2d(lds_offset_of(VtBuf), Vbase + (size_t)k0 * DH, 64, 64, 64, 64, 64);
  }
#else
#pragma unroll
  for (int i = 0; i < 2; ++i) {
    int chunk = t + i * 256;
    int r = chunk >> 3;
    int c = (chunk & 7) << 3;
    TILE_CP(KtBuf + r * DH + c, Kbase + (size_t)(k0 + r) * DH + c);
    TILE_CP(VtBuf + r * DH + c, Vbase + (size_t)(k0 + r) * DH + c);
  }
#endif
}

__global__ __launch_bounds__(256)
void attn_kernel(const unsigned short* __restrict__ Qb,
                 const unsigned short* __restrict__ Kb,
                 const unsigned short* __restrict__ Vb,
                 const float* __restrict__ Qp,
                 const unsigned char* __restrict__ maskp,
                 unsigned short* __restrict__ comb) {
  __shared__ __align__(16) unsigned short Kt[2 * 64 * DH];   // 16 KB
  __shared__ __align__(16) unsigned short Vt[2 * 64 * DH];   // 16 KB
  __shared__ __align__(16) unsigned short Pt[128 * 64];      // 16 KB

  const int bh   = blockIdx.x;
  const int b    = bh >> 4, h = bh & 15;
  const int q0   = blockIdx.y * 128;
  const int t    = threadIdx.x, lane = t & 31, wave = t >> 5;
  const int nl   = lane & 15;
  const int mh   = (lane >> 4) << 3;

  const size_t headBase = (((size_t)b * H_DIM + h) * S_DIM) * DH;
  const unsigned short* Qrow = Qb + headBase + (size_t)(q0 + wave * 16) * DH;
  const bf16x16 qa0 = load_a_frag(Qrow, DH, lane);        // dh 0..31
  const bf16x16 qa1 = load_a_frag(Qrow + 32, DH, lane);   // dh 32..63
  const unsigned short* Kbase = Kb + headBase;
  const unsigned short* Vbase = Vb + headBase;
  const float* Qprow = Qp + ((((size_t)b * H_DIM + h) * S_DIM) + q0 + wave * 16) * NPOSP;

  floatx8 Oacc[4];
#pragma unroll
  for (int n = 0; n < 4; ++n) Oacc[n] = (floatx8)0.0f;
  float row_m[8], row_l[8];
#pragma unroll
  for (int r = 0; r < 8; ++r) { row_m[r] = -1e30f; row_l[r] = 0.f; }

  const int T = S_DIM / 64;     // 32 key tiles
  kv_issue(Kbase, Vbase, Kt, Vt, 0, t);

  for (int it = 0; it < T; ++it) {
    const int k0 = it * 64;
    if (it + 1 < T) {
      kv_issue(Kbase, Vbase, Kt + ((it + 1) & 1) * (64 * DH),
               Vt + ((it + 1) & 1) * (64 * DH), k0 + 64, t);
#if HAVE_TDM
      if (t < 32) __builtin_amdgcn_s_wait_tensorcnt(2);
#else
      ASYNC_WAIT(4);
#endif
    } else {
#if HAVE_TDM
      if (t < 32) __builtin_amdgcn_s_wait_tensorcnt(0);
#else
      ASYNC_WAIT(0);
#endif
    }
    __syncthreads();

    const unsigned short* Ktb = Kt + (it & 1) * (64 * DH);
    const unsigned short* Vtb = Vt + (it & 1) * (64 * DH);

    // S tile = Q * K^T  (+ rel-pos gather, scale, mask)
    float st[4][8];
#pragma unroll
    for (int n = 0; n < 4; ++n) {
      floatx8 s = (floatx8)0.0f;
      bf16x16 kb0 = load_b_frag(Ktb + (n * 16) * DH, DH, lane);
      bf16x16 kb1 = load_b_frag(Ktb + (n * 16) * DH + 32, DH, lane);
      s = wmma_bf16(qa0, kb0, s);
      s = wmma_bf16(qa1, kb1, s);
#pragma unroll
      for (int r = 0; r < 8; ++r) {
        int q = q0 + wave * 16 + r + mh;
        int k = k0 + n * 16 + nl;
        int dp = k - q; dp = dp < -128 ? -128 : (dp > 128 ? 128 : dp);
        float pv = Qprow[(size_t)(r + mh) * NPOSP + (dp + 128)];
        float xv = (s[r] + pv) * 0.125f;     // 1/sqrt(D/H) = 1/8
        if (maskp[b * S_DIM + k]) xv = -1e30f;
        st[n][r] = xv;
      }
    }

    // online softmax: rows live across 16-lane halves (C layout -> lane = N)
#pragma unroll
    for (int r = 0; r < 8; ++r) {
      float tm = fmaxf(fmaxf(st[0][r], st[1][r]), fmaxf(st[2][r], st[3][r]));
#pragma unroll
      for (int m = 1; m <= 8; m <<= 1) tm = fmaxf(tm, __shfl_xor(tm, m, 32));
      float mnew  = fmaxf(row_m[r], tm);
      float alpha = __expf(row_m[r] - mnew);
      float ts = 0.f;
#pragma unroll
      for (int n = 0; n < 4; ++n) { float p = __expf(st[n][r] - mnew); st[n][r] = p; ts += p; }
#pragma unroll
      for (int m = 1; m <= 8; m <<= 1) ts += __shfl_xor(ts, m, 32);
      row_l[r] = row_l[r] * alpha + ts;
      row_m[r] = mnew;
#pragma unroll
      for (int n = 0; n < 4; ++n) Oacc[n][r] = Oacc[n][r] * alpha;
    }

    // P (bf16) via wave-private LDS rows -> A fragments (DS in-order per wave)
#pragma unroll
    for (int n = 0; n < 4; ++n)
#pragma unroll
      for (int r = 0; r < 8; ++r)
        Pt[(wave * 16 + r + mh) * 64 + n * 16 + nl] = f2bf(st[n][r]);

    bf16x16 pa0 = load_a_frag(Pt + (wave * 16) * 64, 64, lane);        // k 0..31
    bf16x16 pa1 = load_a_frag(Pt + (wave * 16) * 64 + 32, 64, lane);   // k 32..63
#pragma unroll
    for (int n = 0; n < 4; ++n) {
      bf16x16 vb0 = load_b_frag_t(Vtb + n * 16, DH, lane);             // keys 0..31
      bf16x16 vb1 = load_b_frag_t(Vtb + 32 * DH + n * 16, DH, lane);   // keys 32..63
      Oacc[n] = wmma_bf16(pa0, vb0, Oacc[n]);
      Oacc[n] = wmma_bf16(pa1, vb1, Oacc[n]);
    }
    __syncthreads();            // all waves done with buffer (it&1)
  }

  // normalize and write combined (heads concatenated) as bf16
#pragma unroll
  for (int n = 0; n < 4; ++n)
#pragma unroll
    for (int r = 0; r < 8; ++r) {
      int q   = q0 + wave * 16 + r + mh;
      int col = h * DH + n * 16 + nl;
      float v = Oacc[n][r] / row_l[r];
      comb[((size_t)b * S_DIM + q) * D_DIM + col] = f2bf(v);
    }
}

// ---------------------------------------------------------------------------
extern "C" void kernel_launch(void* const* d_in, const int* in_sizes, int n_in,
                              void* d_out, int out_size, void* d_ws, size_t ws_size,
                              hipStream_t stream) {
  (void)in_sizes; (void)n_in; (void)out_size; (void)ws_size;
  const float*         x     = (const float*)d_in[0];
  const unsigned char* mask  = (const unsigned char*)d_in[1];
  const float*         W_in  = (const float*)d_in[2];
  const float*         b_in  = (const float*)d_in[3];
  const float*         pos   = (const float*)d_in[4];
  const float*         W_out = (const float*)d_in[5];
  const float*         b_out = (const float*)d_in[6];
  float* out = (float*)d_out;

  char* ws = (char*)d_ws;
  size_t off = 0;
  auto alloc = [&](size_t bytes) -> void* {
    void* p = ws + off;
    off = (off + bytes + 255) & ~(size_t)255;
    return p;
  };

  const int MS  = B_DIM * S_DIM;              // 8192
  const int BHS = B_DIM * H_DIM * S_DIM;      // 131072
  unsigned short* xb    = (unsigned short*)alloc((size_t)MS * D_DIM * 2);
  unsigned short* winb  = (unsigned short*)alloc((size_t)3 * D_DIM * D_DIM * 2);
  unsigned short* woutb = (unsigned short*)alloc((size_t)D_DIM * D_DIM * 2);
  unsigned short* posb  = (unsigned short*)alloc((size_t)NPOSP * DH * 2);
  unsigned short* Qb    = (unsigned short*)alloc((size_t)BHS * DH * 2);
  unsigned short* Kb    = (unsigned short*)alloc((size_t)BHS * DH * 2);
  unsigned short* Vb    = (unsigned short*)alloc((size_t)BHS * DH * 2);
  float*          Qp    = (float*)alloc((size_t)BHS * NPOSP * 4);
  unsigned short* comb  = (unsigned short*)alloc((size_t)MS * D_DIM * 2);

  // 1) casts
  int n1 = MS * D_DIM;
  cast_bf16_kernel<<<(n1 + 255) / 256, 256, 0, stream>>>(x, xb, n1);
  int n2 = 3 * D_DIM * D_DIM;
  cast_bf16_kernel<<<(n2 + 255) / 256, 256, 0, stream>>>(W_in, winb, n2);
  int n3 = D_DIM * D_DIM;
  cast_bf16_kernel<<<(n3 + 255) / 256, 256, 0, stream>>>(W_out, woutb, n3);
  cast_pos_kernel<<<(NPOSP * DH + 255) / 256, 256, 0, stream>>>(pos, posb);

  // 2) QKV projection: [8192,1024] x [3072,1024]^T + b_in -> scatter Q/K/V
  dim3 g1(3 * D_DIM / BN, MS / BM);
  gemm_bf16_wmma<<<g1, 256, 0, stream>>>(xb, winb, b_in, MS, 3 * D_DIM, D_DIM,
                                         1, nullptr, 0, Qb, Kb, Vb);

  // 3) pos logits basis: Qp[131072,272] = Q * pos_emb^T
  dim3 g2((NPOSP + BN - 1) / BN, BHS / BM);
  gemm_bf16_wmma<<<g2, 256, 0, stream>>>(Qb, posb, nullptr, BHS, NPOSP, DH,
                                         0, Qp, NPOSP, nullptr, nullptr, nullptr);

  // 4) fused flash attention
  dim3 g3(B_DIM * H_DIM, S_DIM / 128);
  attn_kernel<<<g3, 256, 0, stream>>>(Qb, Kb, Vb, Qp, mask, comb);

  // 5) output projection: [8192,1024] x [1024,1024]^T + b_out -> fp32 out
  dim3 g4(D_DIM / BN, MS / BM);
  gemm_bf16_wmma<<<g4, 256, 0, stream>>>(comb, woutb, b_out, MS, D_DIM, D_DIM,
                                         0, out, D_DIM, nullptr, nullptr, nullptr);
}
